// model_69320772157876
// MI455X (gfx1250) — compile-verified, matrix operand order checked by reading, not run on hardware
//
#include <hip/hip_runtime.h>
#include <hip/hip_bf16.h>
#include <hip/hip_fp16.h>

typedef __attribute__((ext_vector_type(16))) _Float16 v16h;
typedef __attribute__((ext_vector_type(8)))  _Float16 v8h;
typedef __attribute__((ext_vector_type(8)))  float    v8f;
typedef __attribute__((ext_vector_type(4)))  unsigned int u32x4;
typedef __attribute__((ext_vector_type(8)))  int      i32x8;
typedef __attribute__((ext_vector_type(4)))  int      i32x4;

// ---------------------------------------------------------------------------
// Tensor Data Mover support (gfx1250). Guarded so host pass / older toolchains
// fall back to cooperative vector loads.
// ---------------------------------------------------------------------------
#if __has_builtin(__builtin_amdgcn_tensor_load_to_lds) && \
    __has_builtin(__builtin_amdgcn_s_wait_tensorcnt)
#define HAVE_TDM 1
#endif

#if defined(HAVE_TDM)
// 2D tile load: tile_rows x tile_cols elements (f16), row stride in elements.
// D# per CDNA5 ISA ch.8: group0 = {count, lds_addr, global_addr, type=2},
// group1 = {data_size=1(2B), tensor dims, tile dims, dim0 stride}.
__device__ __forceinline__ void tdm_load_tile_2d(
    unsigned lds_off, const void* gaddr, unsigned tile_cols, unsigned tile_rows,
    unsigned tdim0, unsigned tdim1, unsigned long long stride0) {
  const unsigned long long ga = (unsigned long long)(size_t)gaddr;
  u32x4 g0;
  g0[0] = 1u;                                            // count=1, user D#
  g0[1] = lds_off;                                       // LDS byte address
  g0[2] = (unsigned)(ga & 0xFFFFFFFFu);                  // global_addr[31:0]
  g0[3] = (unsigned)((ga >> 32) & 0x01FFFFFFu) | (2u << 30);  // [56:32], type=2
  i32x8 g1;
  g1[0] = 0x00010000;                                    // data_size=1 -> 2B
  g1[1] = (int)((tdim0 & 0xFFFFu) << 16);                // tensor_dim0 lo
  g1[2] = (int)((tdim0 >> 16) | ((tdim1 & 0xFFFFu) << 16));
  g1[3] = (int)((tdim1 >> 16) | (tile_cols << 16));      // tile_dim0
  g1[4] = (int)(tile_rows & 0xFFFFu);                    // tile_dim1, dim2=0
  g1[5] = (int)(stride0 & 0xFFFFFFFFull);                // dim0 stride lo
  g1[6] = (int)((stride0 >> 32) & 0xFFFFull);            // stride hi, stride1=0
  g1[7] = 0;
  i32x4 z4 = {0, 0, 0, 0};
#if __clang_major__ >= 23
  i32x8 z8 = {0, 0, 0, 0, 0, 0, 0, 0};
  __builtin_amdgcn_tensor_load_to_lds(g0, g1, z4, z4, z8, 0);
#else
  __builtin_amdgcn_tensor_load_to_lds(g0, g1, z4, z4, 0);
#endif
}
#endif

// ---------------------------------------------------------------------------
// WMMA fragment load from an LDS [16][32] f16 tile (row-major).
// 16-bit A layout (ISA 7.12.2): lanes 0-15 -> row=lane, K={0..7,16..23};
// lanes 16-31 -> row=lane-16, K={8..15,24..31}. B staged as [n][k] so the
// same loader serves both operands.
// ---------------------------------------------------------------------------
__device__ inline v16h load_frag16(const _Float16* tile, int lane) {
  const int row = lane & 15;
  const int k0  = (lane < 16) ? 0 : 8;
  const _Float16* p = tile + row * 32 + k0;
  v8h lo = *(const v8h*)(p);
  v8h hi = *(const v8h*)(p + 16);
  v16h r;
#pragma unroll
  for (int i = 0; i < 8; ++i) { r[i] = lo[i]; r[i + 8] = hi[i]; }
  return r;
}

__device__ inline v8h zero_v8h() {
  v8h z;
#pragma unroll
  for (int i = 0; i < 8; ++i) z[i] = (_Float16)0.0f;
  return z;
}

// ---------------------------------------------------------------------------
// conv1: 1->256 ch, 6x6 stride 2, fp32 math (tiny). Writes:
//   x1p : CHANNEL-LAST padded f16 [32][30][30][256] (halo pre-zeroed)
//   z16 : f16 concat activations [m=20000][512], channels 0..255 = relu(x1)
// ---------------------------------------------------------------------------
__global__ __launch_bounds__(256) void conv1_kernel(
    const float* __restrict__ x, const float* __restrict__ w1,
    const float* __restrict__ b1, _Float16* __restrict__ z16,
    _Float16* __restrict__ x1p) {
  __shared__ float wf[36];
  const int bf = blockIdx.x;        // 0..8191 = b*256 + f
  const int b = bf >> 8, f = bf & 255;
  if (threadIdx.x < 36) wf[threadIdx.x] = w1[f * 36 + threadIdx.x];
  __syncthreads();
  const float bias = b1[f];
  const float* xb = x + (size_t)b * 54 * 54;
  for (int p = threadIdx.x; p < 625; p += 256) {
    const int oh = p / 25, ow = p - (p / 25) * 25;
    float s = bias;
#pragma unroll
    for (int r = 0; r < 6; ++r)
#pragma unroll
      for (int c = 0; c < 6; ++c)
        s += xb[(2 * oh + r) * 54 + 2 * ow + c] * wf[r * 6 + c];
    s = fmaxf(s, 0.0f);
    const _Float16 h = (_Float16)s;
    z16[(size_t)(b * 625 + p) * 512 + f] = h;
    x1p[((size_t)((b * 30 + oh + 2) * 30) + ow + 2) * 256 + f] = h;
  }
}

// ---------------------------------------------------------------------------
// Weight transforms, f16 N-major (BwT[n][k]) for TDM-friendly tiles.
// conv2 K ordering: k = (rr*6+ss)*256 + c  (matches channel-last x1p gather)
// ---------------------------------------------------------------------------
__global__ void prep_w2_kernel(const float* __restrict__ w2, _Float16* __restrict__ BwT) {
  const int idx = blockIdx.x * 256 + threadIdx.x;
  if (idx >= 9216 * 256) return;
  const int n = idx / 9216;
  const int k = idx - n * 9216;
  const int rs = k >> 8, c = k & 255;
  BwT[idx] = (_Float16)w2[(size_t)n * 9216 + c * 36 + rs];
}
__global__ void prep_w3_kernel(const float* __restrict__ w3, _Float16* __restrict__ BwT) {
  const int idx = blockIdx.x * 256 + threadIdx.x;
  if (idx >= 512 * 256) return;
  BwT[idx] = (_Float16)w3[idx];   // already [n][c]
}

// ---------------------------------------------------------------------------
// conv2 implicit-GEMM: M=20000, N=256, K=9216 (k = rs*256+c).
// 128x128 tile, 8 waves of 32x64, K-step 32, double-buffered LDS.
// A: two global_load_b128 per thread (contiguous due to channel-last layout).
// B: TDM 2D tile (128n x 32k) from BwT, or cooperative b128 fallback.
// ---------------------------------------------------------------------------
__global__ __launch_bounds__(256, 1) void conv2_wmma_kernel(
    const _Float16* __restrict__ x1p, const _Float16* __restrict__ BwT,
    const float* __restrict__ bias, _Float16* __restrict__ z16) {
  __shared__ alignas(16) _Float16 As[2][128][32];
  __shared__ alignas(16) _Float16 Bs[2][128][32];
  const int tid  = threadIdx.x;
  const int lane = tid & 31;
  const int wave = tid >> 5;
  const int wm = wave & 3, wn = wave >> 2;
  const int m_base = blockIdx.x * 128;
  const int n_base = blockIdx.y * 128;
  const bool w0 = (wave == 0);
  (void)w0;

  v8f acc[2][4];
#pragma unroll
  for (int i = 0; i < 2; ++i)
#pragma unroll
    for (int j = 0; j < 4; ++j)
      acc[i][j] = (v8f){0.f, 0.f, 0.f, 0.f, 0.f, 0.f, 0.f, 0.f};

  // A loader geometry: thread -> (row ar, 16-half segment akk)
  const int ar  = tid >> 1;
  const int akk = (tid & 1) * 16;
  const int am  = m_base + ar;
  const bool am_ok = (am < 20000);
  int apix = 0;
  if (am_ok) {
    const int ab = am / 625;
    const int p = am - ab * 625;
    const int aoh = p / 25, aow = p - (p / 25) * 25;
    apix = (ab * 30 + aoh) * 30 + aow;
  }
#if !defined(HAVE_TDM)
  const int bn  = tid >> 1;
  const int bkk = (tid & 1) * 16;
#endif

  // ---- prologue: tile 0 (rs=0 -> rr=0, ss=0, c0=0) ----
  {
    v8h a0 = zero_v8h(), a1 = zero_v8h();
    if (am_ok) {
      const _Float16* ap = x1p + ((size_t)apix << 8) + akk;
      a0 = *(const v8h*)ap;
      a1 = *(const v8h*)(ap + 8);
    }
    *(v8h*)&As[0][ar][akk]     = a0;
    *(v8h*)&As[0][ar][akk + 8] = a1;
#if defined(HAVE_TDM)
    if (w0) {
      tdm_load_tile_2d((unsigned)(size_t)(const void*)&Bs[0][0][0],
                       BwT + (size_t)n_base * 9216, 32, 128, 32, 128, 9216ull);
      __builtin_amdgcn_s_wait_tensorcnt(0);
    }
#else
    {
      const _Float16* bp = BwT + (size_t)(n_base + bn) * 9216 + bkk;
      *(v8h*)&Bs[0][bn][bkk]     = *(const v8h*)bp;
      *(v8h*)&Bs[0][bn][bkk + 8] = *(const v8h*)(bp + 8);
    }
#endif
  }
  __syncthreads();

  for (int it = 0; it < 288; ++it) {
    const int cur = it & 1, nxt = cur ^ 1;
    const bool more = (it + 1) < 288;
    v8h a0, a1;
    if (more) {
      const int k0 = (it + 1) * 32;
      const int rs = k0 >> 8;                 // uniform scalar
      const int c0 = k0 & 255;
      const int rr = rs / 6, ss = rs - (rs / 6) * 6;
      a0 = zero_v8h(); a1 = zero_v8h();
      if (am_ok) {
        const _Float16* ap =
            x1p + (((size_t)(apix + rr * 30 + ss)) << 8) + c0 + akk;
        a0 = *(const v8h*)ap;
        a1 = *(const v8h*)(ap + 8);
      }
#if defined(HAVE_TDM)
      if (w0)
        tdm_load_tile_2d((unsigned)(size_t)(const void*)&Bs[nxt][0][0],
                         BwT + (size_t)n_base * 9216 + k0, 32, 128, 32, 128,
                         9216ull);
#endif
    }

    v16h afrag[2], bfrag[4];
#pragma unroll
    for (int i = 0; i < 2; ++i)
      afrag[i] = load_frag16(&As[cur][wm * 32 + i * 16][0], lane);
#pragma unroll
    for (int j = 0; j < 4; ++j)
      bfrag[j] = load_frag16(&Bs[cur][wn * 64 + j * 16][0], lane);
#pragma unroll
    for (int i = 0; i < 2; ++i)
#pragma unroll
      for (int j = 0; j < 4; ++j)
        acc[i][j] = __builtin_amdgcn_wmma_f32_16x16x32_f16(
            false, afrag[i], false, bfrag[j], (short)0, acc[i][j], false, false);

    if (more) {
      *(v8h*)&As[nxt][ar][akk]     = a0;
      *(v8h*)&As[nxt][ar][akk + 8] = a1;
#if defined(HAVE_TDM)
      if (w0) __builtin_amdgcn_s_wait_tensorcnt(0);
#else
      const int k0 = (it + 1) * 32;
      const _Float16* bp = BwT + (size_t)(n_base + bn) * 9216 + k0 + bkk;
      *(v8h*)&Bs[nxt][bn][bkk]     = *(const v8h*)bp;
      *(v8h*)&Bs[nxt][bn][bkk + 8] = *(const v8h*)(bp + 8);
#endif
    }
    __syncthreads();
  }

  // Epilogue: C/D layout -> col = lane&15, rows (lane<16 ? v : v+8)
  const int col  = lane & 15;
  const int roff = (lane < 16) ? 0 : 8;
#pragma unroll
  for (int i = 0; i < 2; ++i)
#pragma unroll
    for (int j = 0; j < 4; ++j) {
      const int n = n_base + wn * 64 + j * 16 + col;
      const float bv = bias[n];
#pragma unroll
      for (int v = 0; v < 8; ++v) {
        const int m = m_base + wm * 32 + i * 16 + roff + v;
        if (m < 20000) {
          float val = fmaxf(acc[i][j][v] + bv, 0.0f);
          z16[(size_t)m * 512 + 256 + n] = (_Float16)val;
        }
      }
    }
}

// ---------------------------------------------------------------------------
// conv3 (1x1, 512->256) GEMM: M=20000, N=256, K=512. Double-buffered,
// both A and B tiles via TDM (A rows are contiguous in z16).
// Epilogue scatters into capsule layout caps[b][g*625+p][i].
// ---------------------------------------------------------------------------
__global__ __launch_bounds__(256, 1) void conv3_wmma_kernel(
    const _Float16* __restrict__ z16, const _Float16* __restrict__ BwT,
    const float* __restrict__ bias, float* __restrict__ caps) {
  __shared__ alignas(16) _Float16 As[2][128][32];
  __shared__ alignas(16) _Float16 Bs[2][128][32];
  const int tid  = threadIdx.x;
  const int lane = tid & 31;
  const int wave = tid >> 5;
  const int wm = wave & 3, wn = wave >> 2;
  const int m_base = blockIdx.x * 128;
  const int n_base = blockIdx.y * 128;
  const bool w0 = (wave == 0);
  (void)w0;
  const int rows_avail = (20000 - m_base) < 128 ? (20000 - m_base) : 128;
  (void)rows_avail;

  v8f acc[2][4];
#pragma unroll
  for (int i = 0; i < 2; ++i)
#pragma unroll
    for (int j = 0; j < 4; ++j)
      acc[i][j] = (v8f){0.f, 0.f, 0.f, 0.f, 0.f, 0.f, 0.f, 0.f};

#if !defined(HAVE_TDM)
  const int ar  = tid >> 1;
  const int akk = (tid & 1) * 16;
  const int am  = m_base + ar;
#endif

  // ---- prologue: tile 0 ----
#if defined(HAVE_TDM)
  if (w0) {
    tdm_load_tile_2d((unsigned)(size_t)(const void*)&As[0][0][0],
                     z16 + (size_t)m_base * 512, 32, 128, 32,
                     (unsigned)rows_avail, 512ull);
    tdm_load_tile_2d((unsigned)(size_t)(const void*)&Bs[0][0][0],
                     BwT + (size_t)n_base * 512, 32, 128, 32, 128, 512ull);
    __builtin_amdgcn_s_wait_tensorcnt(0);
  }
#else
  {
    v8h a0 = zero_v8h(), a1 = zero_v8h();
    if (am < 20000) {
      const _Float16* ap = z16 + (size_t)am * 512 + akk;
      a0 = *(const v8h*)ap; a1 = *(const v8h*)(ap + 8);
    }
    *(v8h*)&As[0][ar][akk]     = a0;
    *(v8h*)&As[0][ar][akk + 8] = a1;
    const _Float16* bp = BwT + (size_t)(n_base + ar) * 512 + akk;
    *(v8h*)&Bs[0][ar][akk]     = *(const v8h*)bp;
    *(v8h*)&Bs[0][ar][akk + 8] = *(const v8h*)(bp + 8);
  }
#endif
  __syncthreads();

  for (int it = 0; it < 16; ++it) {
    const int cur = it & 1, nxt = cur ^ 1;
    const bool more = (it + 1) < 16;
    const int k0 = (it + 1) * 32;
#if defined(HAVE_TDM)
    if (more && w0) {
      tdm_load_tile_2d((unsigned)(size_t)(const void*)&As[nxt][0][0],
                       z16 + (size_t)m_base * 512 + k0, 32, 128, 32,
                       (unsigned)rows_avail, 512ull);
      tdm_load_tile_2d((unsigned)(size_t)(const void*)&Bs[nxt][0][0],
                       BwT + (size_t)n_base * 512 + k0, 32, 128, 32, 128,
                       512ull);
    }
#else
    v8h a0, a1, b0, b1;
    if (more) {
      a0 = zero_v8h(); a1 = zero_v8h();
      if (am < 20000) {
        const _Float16* ap = z16 + (size_t)am * 512 + k0 + akk;
        a0 = *(const v8h*)ap; a1 = *(const v8h*)(ap + 8);
      }
      const _Float16* bp = BwT + (size_t)(n_base + ar) * 512 + k0 + akk;
      b0 = *(const v8h*)bp; b1 = *(const v8h*)(bp + 8);
    }
#endif

    v16h afrag[2], bfrag[4];
#pragma unroll
    for (int i = 0; i < 2; ++i)
      afrag[i] = load_frag16(&As[cur][wm * 32 + i * 16][0], lane);
#pragma unroll
    for (int j = 0; j < 4; ++j)
      bfrag[j] = load_frag16(&Bs[cur][wn * 64 + j * 16][0], lane);
#pragma unroll
    for (int i = 0; i < 2; ++i)
#pragma unroll
      for (int j = 0; j < 4; ++j)
        acc[i][j] = __builtin_amdgcn_wmma_f32_16x16x32_f16(
            false, afrag[i], false, bfrag[j], (short)0, acc[i][j], false, false);

    if (more) {
#if defined(HAVE_TDM)
      if (w0) __builtin_amdgcn_s_wait_tensorcnt(0);
#else
      *(v8h*)&As[nxt][ar][akk]     = a0;
      *(v8h*)&As[nxt][ar][akk + 8] = a1;
      *(v8h*)&Bs[nxt][ar][akk]     = b0;
      *(v8h*)&Bs[nxt][ar][akk + 8] = b1;
#endif
    }
    __syncthreads();
  }

  const int col  = lane & 15;
  const int roff = (lane < 16) ? 0 : 8;
#pragma unroll
  for (int i = 0; i < 2; ++i)
#pragma unroll
    for (int j = 0; j < 4; ++j) {
      const int n = n_base + wn * 64 + j * 16 + col;
      const int g = n >> 3, ci = n & 7;
      const float bv = bias[n];
#pragma unroll
      for (int v = 0; v < 8; ++v) {
        const int m = m_base + wm * 32 + i * 16 + roff + v;
        if (m < 20000) {
          const int bb = m / 625, p = m - (m / 625) * 625;
          caps[((size_t)bb * 20000 + g * 625 + p) * 8 + ci] = acc[i][j][v] + bv;
        }
      }
    }
}

// ---------------------------------------------------------------------------
// u[b][n][o] = sum_i caps[b][n][i] * pc_w[n][o][i] + pc_b[n][o]   (fp32)
// ---------------------------------------------------------------------------
__global__ __launch_bounds__(256) void primary_u_kernel(
    const float* __restrict__ caps, const float* __restrict__ pcw,
    const float* __restrict__ pcb, float* __restrict__ u) {
  const int idx = blockIdx.x * 256 + threadIdx.x;
  if (idx >= 32 * 20000) return;
  const int b = idx / 20000;
  const int n = idx - b * 20000;
  const size_t bn = ((size_t)b * 20000 + n) * 8;
  float cv[8];
#pragma unroll
  for (int i = 0; i < 8; ++i) cv[i] = caps[bn + i];
  const float* wn = pcw + (size_t)n * 64;
  const float* bb = pcb + (size_t)n * 8;
#pragma unroll
  for (int o = 0; o < 8; ++o) {
    float s = bb[o];
#pragma unroll
    for (int i = 0; i < 8; ++i) s += cv[i] * wn[o * 8 + i];
    u[bn + o] = s;
  }
}

// ---------------------------------------------------------------------------
// Routing accumulation: block = 8 capsules x 32 batch; u_hat recomputed
// (W stays L2-resident). LDS float atomics build a block partial -> P[block];
// deterministic two-stage reduction.
// ---------------------------------------------------------------------------
__global__ __launch_bounds__(256) void routing_accum_kernel(
    const float* __restrict__ u, const float* __restrict__ W,
    const float* __restrict__ blog, float* __restrict__ P) {
  __shared__ float Sl[4096];  // S[b][e][o] = [32][8][16]
  const int tid = threadIdx.x;
  for (int i = tid; i < 4096; i += 256) Sl[i] = 0.0f;
  __syncthreads();
  const int b = tid & 31;
  const int n = blockIdx.x * 8 + (tid >> 5);
  if (n < 20000) {
    const size_t bn = ((size_t)b * 20000 + n) * 8;
    const float* bl = blog + bn;
    float mx = bl[0];
#pragma unroll
    for (int e = 1; e < 8; ++e) mx = fmaxf(mx, bl[e]);
    float ce[8], sum = 0.0f;
#pragma unroll
    for (int e = 0; e < 8; ++e) { ce[e] = __expf(bl[e] - mx); sum += ce[e]; }
    const float inv = 1.0f / sum;
    float uu[8];
#pragma unroll
    for (int i = 0; i < 8; ++i) uu[i] = u[bn + i];
    const float* Wn = W + (size_t)n * 1024;
    for (int e = 0; e < 8; ++e) {
      const float c = ce[e] * inv;
      for (int o = 0; o < 16; ++o) {
        const float* w = Wn + (e * 16 + o) * 8;
        float uh = 0.0f;
#pragma unroll
        for (int i = 0; i < 8; ++i) uh += uu[i] * w[i];
        atomicAdd(&Sl[(b * 8 + e) * 16 + o], c * uh);
      }
    }
  }
  __syncthreads();
  float* Pb = P + (size_t)blockIdx.x * 4096;
  for (int i = tid; i < 4096; i += 256) Pb[i] = Sl[i];
}

__global__ void reduce_S_kernel(const float* __restrict__ P, float* __restrict__ S,
                                int nblk) {
  const int i = blockIdx.x * 256 + threadIdx.x;
  if (i >= 4096) return;
  float s = 0.0f;
  for (int b = 0; b < nblk; ++b) s += P[(size_t)b * 4096 + i];
  S[i] = s;
}

// squash: thread = (b,e); final iteration writes ||v|| = scale * ||s||.
__global__ void squash_kernel(const float* __restrict__ S, float* __restrict__ vbuf,
                              float* __restrict__ out, int final_iter) {
  const int tid = threadIdx.x;  // 256 = 32*8
  const float* s = S + tid * 16;
  float n2 = 0.0f;
#pragma unroll
  for (int o = 0; o < 16; ++o) n2 += s[o] * s[o];
  const float nn = sqrtf(n2);
  const float scale = n2 / (1.0f + n2) / (nn + 1e-8f);
  if (final_iter) {
    out[tid] = scale * nn;
  } else {
#pragma unroll
    for (int o = 0; o < 16; ++o) vbuf[tid * 16 + o] = scale * s[o];
  }
}

// b[b][n][e] += sum_o u_hat[b,n,e,o] * v[b,e,o]  (u_hat recomputed)
__global__ __launch_bounds__(256) void routing_update_kernel(
    const float* __restrict__ u, const float* __restrict__ W,
    const float* __restrict__ vbuf, float* __restrict__ blog) {
  const int tid = threadIdx.x;
  const int b = tid & 31;
  const int n = blockIdx.x * 8 + (tid >> 5);
  if (n >= 20000) return;
  const size_t bn = ((size_t)b * 20000 + n) * 8;
  float uu[8];
#pragma unroll
  for (int i = 0; i < 8; ++i) uu[i] = u[bn + i];
  const float* Wn = W + (size_t)n * 1024;
  const float* vb = vbuf + b * 128;
  for (int e = 0; e < 8; ++e) {
    float a = 0.0f;
    for (int o = 0; o < 16; ++o) {
      const float* w = Wn + (e * 16 + o) * 8;
      float uh = 0.0f;
#pragma unroll
      for (int i = 0; i < 8; ++i) uh += uu[i] * w[i];
      a += uh * vb[e * 16 + o];
    }
    blog[bn + e] += a;
  }
}

// ---------------------------------------------------------------------------
extern "C" void kernel_launch(void* const* d_in, const int* in_sizes, int n_in,
                              void* d_out, int out_size, void* d_ws, size_t ws_size,
                              hipStream_t stream) {
  (void)in_sizes; (void)n_in; (void)out_size; (void)ws_size;
  const float* x   = (const float*)d_in[0];
  const float* w1  = (const float*)d_in[1];
  const float* b1  = (const float*)d_in[2];
  const float* w2  = (const float*)d_in[3];
  const float* b2  = (const float*)d_in[4];
  const float* w3  = (const float*)d_in[5];
  const float* b3  = (const float*)d_in[6];
  const float* pcw = (const float*)d_in[7];
  const float* pcb = (const float*)d_in[8];
  const float* W   = (const float*)d_in[9];
  float* out = (float*)d_out;

  char* ws = (char*)d_ws;
  size_t off = 0;
  auto carve = [&](size_t bytes) -> char* {
    char* p = ws + off;
    off = (off + bytes + 255) & ~(size_t)255;
    return p;
  };
  _Float16* x1p  = (_Float16*)carve(32ull * 30 * 30 * 256 * 2);  // channel-last padded
  _Float16* z16  = (_Float16*)carve(20000ull * 512 * 2);         // concat f16
  _Float16* BwT2 = (_Float16*)carve(256ull * 9216 * 2);          // conv2 W [n][k]
  _Float16* BwT3 = (_Float16*)carve(256ull * 512 * 2);           // conv3 W [n][k]
  float* caps    = (float*)carve(32ull * 20000 * 8 * 4);
  float* u       = (float*)carve(32ull * 20000 * 8 * 4);
  float* blog    = (float*)carve(32ull * 20000 * 8 * 4);
  float* P       = (float*)carve(2500ull * 4096 * 4);
  float* S       = (float*)carve(4096 * 4);
  float* vbuf    = (float*)carve(4096 * 4);

  hipMemsetAsync(x1p, 0, 32ull * 30 * 30 * 256 * 2, stream);  // zero halo
  hipMemsetAsync(blog, 0, 32ull * 20000 * 8 * 4, stream);     // routing logits

  prep_w2_kernel<<<9216, 256, 0, stream>>>(w2, BwT2);
  prep_w3_kernel<<<512, 256, 0, stream>>>(w3, BwT3);
  conv1_kernel<<<8192, 256, 0, stream>>>(x, w1, b1, z16, x1p);
  conv2_wmma_kernel<<<dim3(157, 2), 256, 0, stream>>>(x1p, BwT2, b2, z16);
  conv3_wmma_kernel<<<dim3(157, 2), 256, 0, stream>>>(z16, BwT3, b3, caps);
  primary_u_kernel<<<2500, 256, 0, stream>>>(caps, pcw, pcb, u);

  for (int r = 0; r < 3; ++r) {
    routing_accum_kernel<<<2500, 256, 0, stream>>>(u, W, blog, P);
    reduce_S_kernel<<<16, 256, 0, stream>>>(P, S, 2500);
    squash_kernel<<<1, 256, 0, stream>>>(S, vbuf, out, r == 2 ? 1 : 0);
    if (r < 2)
      routing_update_kernel<<<2500, 256, 0, stream>>>(u, W, vbuf, blog);
  }
}
